// Attention_120259084755
// MI455X (gfx1250) — matrix operand order repaired and checked by reading, not compiled, not run on hardware
//
#include <hip/hip_runtime.h>

typedef __attribute__((ext_vector_type(16))) __bf16 v16bf;
typedef __attribute__((ext_vector_type(8)))  __bf16 v8bf;
typedef __attribute__((ext_vector_type(8)))  float  v8f;
typedef __attribute__((ext_vector_type(4)))  unsigned int u32x4;
typedef __attribute__((ext_vector_type(8)))  unsigned int u32x8;

static constexpr int   S_LEN   = 2048;
static constexpr int   D_KEY   = 512;
// (1/sqrt(512)) * log2(e): online softmax runs in the exp2 domain (v_exp_f32 is exp2).
static constexpr float SCL_LOG2 = 0.06375872465402481f;
static constexpr float NEG_BIG  = -1e30f;

// Pack 16 fp32 -> bf16x16 WMMA operand vector (v_cvt_pk_bf16_f32 pairs).
__device__ __forceinline__ v16bf make_bf16x16(float4 a0, float4 a1, float4 b0, float4 b1) {
  v16bf r;
  r[0]  = (__bf16)a0.x; r[1]  = (__bf16)a0.y; r[2]  = (__bf16)a0.z; r[3]  = (__bf16)a0.w;
  r[4]  = (__bf16)a1.x; r[5]  = (__bf16)a1.y; r[6]  = (__bf16)a1.z; r[7]  = (__bf16)a1.w;
  r[8]  = (__bf16)b0.x; r[9]  = (__bf16)b0.y; r[10] = (__bf16)b0.z; r[11] = (__bf16)b0.w;
  r[12] = (__bf16)b1.x; r[13] = (__bf16)b1.y; r[14] = (__bf16)b1.z; r[15] = (__bf16)b1.w;
  return r;
}

// A-operand (16x32 bf16): lane = M, half h: elems 0..7 = K 8h..8h+7, elems 8..15 = K 16+8h..+7
__device__ __forceinline__ v16bf load_A_operand(const float* __restrict__ rowp, int h) {
  const float4* p0 = reinterpret_cast<const float4*>(rowp + 8 * h);
  const float4* p1 = reinterpret_cast<const float4*>(rowp + 16 + 8 * h);
  return make_bf16x16(p0[0], p0[1], p1[0], p1[1]);
}

// B-operand (32x16 bf16): lane = N (rowp = source row for column N), half h: elems 0..7 = K 8h..8h+7, elems 8..15 = K 16+8h..+7
__device__ __forceinline__ v16bf load_B_operand(const float* rowp, int h) {
  const float4* p0 = reinterpret_cast<const float4*>(rowp + 8 * h);
  const float4* p1 = reinterpret_cast<const float4*>(rowp + 16 + 8 * h);
  return make_bf16x16(p0[0], p0[1], p1[0], p1[1]);
}

// TDM: DMA a 32-row x 128-col f32 tile (row stride 512 elems) from global into LDS.
// D# built per CDNA5 ISA ch.8: group0 = ctrl/lds/global addr, group1 = dims, groups2/3 = 0 (2D).
__device__ __forceinline__ void tdm_load_2d(const float* gptr, const float* lptr) {
  unsigned long long ga = (unsigned long long)(uintptr_t)gptr;   // flat VA (57-bit)
  unsigned lds = (unsigned)(uintptr_t)lptr;                      // low 32 bits = LDS byte offset
  u32x4 g0;
  g0[0] = 1u;                                                       // count=1, no gather
  g0[1] = __builtin_amdgcn_readfirstlane(lds);                      // lds_addr [63:32]
  g0[2] = __builtin_amdgcn_readfirstlane((unsigned)ga);             // global_addr lo
  g0[3] = __builtin_amdgcn_readfirstlane(
            (unsigned)(ga >> 32) | (2u << 30));                     // global_addr hi | type=2
  u32x8 g1;
  g1[0] = 2u << 16;                            // wg_mask=0, data_size=2 (4 bytes)
  g1[1] = (unsigned)D_KEY << 16;               // tensor_dim0 lo16 (=512)
  g1[2] = ((unsigned)D_KEY >> 16) | ((unsigned)S_LEN << 16);   // dim0 hi | tensor_dim1 lo (=2048)
  g1[3] = ((unsigned)S_LEN >> 16) | (128u << 16);              // dim1 hi | tile_dim0 = 128
  g1[4] = 32u;                                 // tile_dim1 = 32, tile_dim2 = 0
  g1[5] = (unsigned)D_KEY;                     // tensor_dim0_stride = 512 (lo32)
  g1[6] = 0u;
  g1[7] = 0u;
  u32x4 gz = {0u, 0u, 0u, 0u};
  asm volatile("tensor_load_to_lds %0, %1, %2, %3"
               :: "s"(g0), "s"(g1), "s"(gz), "s"(gz)
               : "memory");
}

__global__ __launch_bounds__(128, 1)
void fa_causal_fwd(const float* __restrict__ Qg, const float* __restrict__ Kg,
                   const float* __restrict__ Vg, float* __restrict__ Og) {
  // Per-wave double-buffered K/V slice staging, filled by the Tensor Data Mover.
  __shared__ float  Kst[4][2][32][128];   // 128 KB
  __shared__ float  Vst[4][2][32][128];   // 128 KB
  __shared__ float  Sred[16][32][4];      // split-K partials: 4 wave values contiguous -> b128 gather
  __shared__ __bf16 Pbuf[4][16][32];      // per-wave P tile, C-layout -> A-layout relay

  const int tid   = threadIdx.x;
  const int lane  = tid & 31;
  const int wave  = tid >> 5;
  const int ln16  = lane & 15;
  const int h     = lane >> 4;       // half: 0 or 1
  const int qblk  = blockIdx.x;      // 0..S/16-1
  const int batch = blockIdx.y;
  const int qbase = qblk * 16;
  const int dbase = wave * 128;      // this wave's D slice

  const float* Qb = Qg + (size_t)batch * S_LEN * D_KEY;
  const float* Kb = Kg + (size_t)batch * S_LEN * D_KEY;
  const float* Vb = Vg + (size_t)batch * S_LEN * D_KEY;

  // Preload Q A-operands for our 128-wide D slice: 4 chunks of K=32 (resident in VGPRs).
  v16bf qa[4];
  {
    const float* qrow = Qb + (size_t)(qbase + ln16) * D_KEY + dbase;
#pragma unroll
    for (int c = 0; c < 4; ++c) qa[c] = load_A_operand(qrow + 32 * c, h);
  }

  v8f oacc[8];
#pragma unroll
  for (int t = 0; t < 8; ++t) oacc[t] = {};
  float mrun[8], lrun[8];
#pragma unroll
  for (int r = 0; r < 8; ++r) { mrun[r] = NEG_BIG; lrun[r] = 0.0f; }

  const int nkb = (qblk >> 1) + 1;   // causal: number of 32-key blocks

  // Prologue: DMA block 0 into buffer 0 (each wave fetches only its own D slice).
  tdm_load_2d(Kb + dbase, &Kst[wave][0][0][0]);
  tdm_load_2d(Vb + dbase, &Vst[wave][0][0][0]);

  for (int j = 0; j < nkb; ++j) {
    const int kb  = j * 32;
    const int cur = j & 1;

    // Pipeline: kick off next block's DMA, then wait for current block's two loads.
    if (j + 1 < nkb) {
      const int nb = cur ^ 1;
      tdm_load_2d(Kb + (size_t)(kb + 32) * D_KEY + dbase, &Kst[wave][nb][0][0]);
      tdm_load_2d(Vb + (size_t)(kb + 32) * D_KEY + dbase, &Vst[wave][nb][0][0]);
      __builtin_amdgcn_s_wait_tensorcnt(2);   // current K,V landed; next two in flight
    } else {
      __builtin_amdgcn_s_wait_tensorcnt(0);
    }

    // ---- 1) partial scores over our D slice: S[16 x 32] += Qslice @ Kslice^T ----
    v8f s0 = {}, s1 = {};
    const float* krow0 = &Kst[wave][cur][ln16][0];        // keys kb..kb+15 (slice-local)
    const float* krow1 = &Kst[wave][cur][16 + ln16][0];   // keys kb+16..kb+31
#pragma unroll
    for (int c = 0; c < 4; ++c) {
      v16bf kop0 = load_B_operand(krow0 + 32 * c, h);     // ds_load_b128 x4
      s0 = __builtin_amdgcn_wmma_f32_16x16x32_bf16(false, qa[c], false, kop0, (short)0, s0, false, false);
      v16bf kop1 = load_B_operand(krow1 + 32 * c, h);
      s1 = __builtin_amdgcn_wmma_f32_16x16x32_bf16(false, qa[c], false, kop1, (short)0, s1, false, false);
    }

    // ---- 2) cross-wave split-K reduction through LDS ----
#pragma unroll
    for (int r = 0; r < 8; ++r) {
      Sred[r + 8 * h][ln16][wave]      = s0[r];
      Sred[r + 8 * h][ln16 + 16][wave] = s1[r];
    }
    __syncthreads();
    float sc0[8], sc1[8];
#pragma unroll
    for (int r = 0; r < 8; ++r) {
      const int m = r + 8 * h;
      const float4 g0 = *reinterpret_cast<const float4*>(&Sred[m][ln16][0]);
      const float4 g1 = *reinterpret_cast<const float4*>(&Sred[m][ln16 + 16][0]);
      sc0[r] = (g0.x + g0.y) + (g0.z + g0.w);
      sc1[r] = (g1.x + g1.y) + (g1.z + g1.w);
    }

    // ---- 3) causal mask + online softmax in exp2 domain (replicated) ----
#pragma unroll
    for (int r = 0; r < 8; ++r) {
      const int mg  = qbase + r + 8 * h;
      const int ng0 = kb + ln16;
      const int ng1 = kb + 16 + ln16;
      float v0 = (ng0 > mg) ? NEG_BIG : sc0[r] * SCL_LOG2;
      float v1 = (ng1 > mg) ? NEG_BIG : sc1[r] * SCL_LOG2;
      float rm = fmaxf(v0, v1);
      rm = fmaxf(rm, __shfl_xor(rm, 1));
      rm = fmaxf(rm, __shfl_xor(rm, 2));
      rm = fmaxf(rm, __shfl_xor(rm, 4));
      rm = fmaxf(rm, __shfl_xor(rm, 8));
      const float mnew  = fmaxf(mrun[r], rm);
      const float alpha = __builtin_amdgcn_exp2f(mrun[r] - mnew);
      mrun[r] = mnew;
      const float p0 = __builtin_amdgcn_exp2f(v0 - mnew);
      const float p1 = __builtin_amdgcn_exp2f(v1 - mnew);
      float rs = p0 + p1;
      rs += __shfl_xor(rs, 1);
      rs += __shfl_xor(rs, 2);
      rs += __shfl_xor(rs, 4);
      rs += __shfl_xor(rs, 8);
      lrun[r] = lrun[r] * alpha + rs;
#pragma unroll
      for (int t = 0; t < 8; ++t) oacc[t][r] = oacc[t][r] * alpha;
      Pbuf[wave][r + 8 * h][ln16]      = (__bf16)p0;
      Pbuf[wave][r + 8 * h][ln16 + 16] = (__bf16)p1;
    }
    __syncthreads(); // release Sred for next iter + order Pbuf write->read

    // ---- 4) read P back in A-operand layout (two ds_load_b128 per lane) ----
    v16bf pa;
    {
      const v8bf lo = *reinterpret_cast<const v8bf*>(&Pbuf[wave][ln16][8 * h]);
      const v8bf hi = *reinterpret_cast<const v8bf*>(&Pbuf[wave][ln16][16 + 8 * h]);
#pragma unroll
      for (int e = 0; e < 8; ++e) { pa[e] = lo[e]; pa[8 + e] = hi[e]; }
    }

    // ---- 5) O[16 x 128slice] += P(16x32) @ V(32x128slice), V from LDS stage ----
#pragma unroll
    for (int t = 0; t < 8; ++t) {
      const float* vcol = &Vst[wave][cur][0][16 * t + ln16];
      v16bf vb;
#pragma unroll
      for (int e = 0; e < 16; ++e) vb[e] = (__bf16)vcol[(8 * h + e + (e >= 8 ? 8 : 0)) * 128];
      oacc[t] = __builtin_amdgcn_wmma_f32_16x16x32_bf16(false, pa, false, vb, (short)0, oacc[t], false, false);
    }
  }

  // ---- epilogue: normalize rows and store ----
  float* orow = Og + ((size_t)batch * S_LEN + qbase) * D_KEY + dbase;
#pragma unroll
  for (int r = 0; r < 8; ++r) {
    const float inv = 1.0f / lrun[r];
    const int m = r + 8 * h;
#pragma unroll
    for (int t = 0; t < 8; ++t)
      orow[(size_t)m * D_KEY + 16 * t + ln16] = oacc[t][r] * inv;
  }
}

extern "C" void kernel_launch(void* const* d_in, const int* in_sizes, int n_in,
                              void* d_out, int out_size, void* d_ws, size_t ws_size,
                              hipStream_t stream) {
  const float* Q = (const float*)d_in[0];
  const float* K = (const float*)d_in[1];
  const float* V = (const float*)d_in[2];
  float* O = (float*)d_out;
  const int B = in_sizes[0] / (S_LEN * D_KEY);
  dim3 grid(S_LEN / 16, B);
  fa_causal_fwd<<<grid, 128, 0, stream>>>(Q, K, V, O);
}
